// VecAttGRU_38646115729851
// MI455X (gfx1250) — matrix operand order repaired and checked by reading, not compiled, not run on hardware
//
#include <hip/hip_runtime.h>
#include <hip/hip_bf16.h>
#include <stdint.h>

// ---------------- problem constants ----------------
#define T_STEPS 200
#define BATCH   1024
#define IDIM    128
#define HDIM    128
#define BTILE   32             // batch rows per workgroup
#define NWG     (BATCH/BTILE)  // 32 workgroups

// padded LDS strides (bank-conflict avoidance; all 16B-aligned)
#define WSTR    264            // bf16 row stride for Wg/Wc/xh (528 B)
#define RHSTR   136            // bf16 row stride for r*h buffer (272 B)
#define FSTR    132            // f32 row stride for h / u buffers

// ---- LDS byte layout (dynamic smem starts at LDS offset 0: no static __shared__) ----
#define OFF_WG     0u
#define OFF_WC     (OFF_WG  + 256u*WSTR*2u)      // 135168
#define OFF_XH     (OFF_WC  + 128u*WSTR*2u)      // 202752
#define OFF_RH     (OFF_XH  + (unsigned)BTILE*WSTR*2u)
#define OFF_H      (OFF_RH  + (unsigned)BTILE*RHSTR*2u)
#define OFF_U      (OFF_H   + (unsigned)BTILE*FSTR*4u)
#define OFF_BG     (OFF_U   + (unsigned)BTILE*FSTR*4u)
#define OFF_BC     (OFF_BG  + 256u*4u)
#define OFF_A      (OFF_BC  + 128u*4u)
#define OFF_SL     (OFF_A   + (unsigned)BTILE*4u)
#define OFF_STAGE  (OFF_SL  + (unsigned)BTILE*4u)
#define STAGE_BYTES ((unsigned)BTILE*IDIM*4u)    // 16 KB per buffer
#define SMEM_BYTES (OFF_STAGE + 2u*STAGE_BYTES)  // ~290 KB < 320 KB/WGP

typedef __attribute__((ext_vector_type(16))) __bf16 v16bf;
typedef __attribute__((ext_vector_type(8)))  float  v8f;
typedef __attribute__((ext_vector_type(4)))  unsigned int u32x4;
typedef __attribute__((ext_vector_type(8)))  int    i32x8;
typedef __attribute__((ext_vector_type(4)))  int    i32x4;

#if defined(__has_builtin)
#if __has_builtin(__builtin_amdgcn_tensor_load_to_lds) && __has_builtin(__builtin_amdgcn_s_wait_tensorcnt)
#define HAVE_TDM 1
#endif
#endif

// A-fragment (16x32 bf16): lane group g=lane>>4 holds K {16g..16g+15}
// -> one contiguous 32-byte chunk of this lane's row.
__device__ __forceinline__ v16bf load_frag_a(const __bf16* rowk, int g) {
    union { uint4 q[2]; v16bf v; } u;
    u.q[0] = *(const uint4*)(rowk + 16*g);
    u.q[1] = *(const uint4*)(rowk + 16*g + 8);
    return u.v;
}
// B-fragment (32x16 bf16): lane holds 16 contiguous bf16 of weight row n = lane&15,
// starting at k-offset 16*(lane>>4)  -> one 32-byte contiguous chunk.
__device__ __forceinline__ v16bf load_frag_b(const __bf16* p) {
    union { uint4 q[2]; v16bf v; } u;
    u.q[0] = *(const uint4*)(p);
    u.q[1] = *(const uint4*)(p + 8);
    return u.v;
}

// rcp-based activations: one v_exp_f32 + one v_rcp_f32, saturate correctly at +-inf
__device__ __forceinline__ float sigmoidf_fast(float x) {
    return __builtin_amdgcn_rcpf(1.0f + __expf(-x));
}
__device__ __forceinline__ float tanhf_fast(float x) {
    return 1.0f - 2.0f * __builtin_amdgcn_rcpf(1.0f + __expf(2.0f * x));
}

#ifdef HAVE_TDM
// Issue one TDM load of a 32x128 f32 tile (row-major, stride 128) into LDS at ldsOff.
// D# per CDNA5 ISA ch.8: group0 = {count|flags, lds_addr, global_addr[56:0], type=2},
// group1 = {mask=0, data_size=4B, dims/strides}. 2D tensor -> groups 2/3 zero (NULL-ish).
__device__ __forceinline__ void tdm_load_x(const float* gsrc, unsigned int ldsOff) {
    unsigned long long ga = (unsigned long long)(uintptr_t)gsrc;
    u32x4 g0;
    g0[0] = 1u;                                               // count=1 (valid), user mode
    g0[1] = ldsOff;                                           // lds_addr (bytes)
    g0[2] = (unsigned int)(ga & 0xffffffffu);                 // global_addr[31:0]
    g0[3] = (unsigned int)((ga >> 32) & 0x01ffffffu) | (2u << 30);  // addr[56:32] | type=2
    i32x8 g1;
    g1[0] = (int)(2u << 16);                 // workgroup_mask=0, data_size=2 (4 bytes)
    g1[1] = (int)((IDIM & 0xffffu) << 16);   // tensor_dim0[15:0] in bits[63:48]
    g1[2] = (int)((IDIM >> 16) | ((BTILE & 0xffffu) << 16)); // dim0 hi | tensor_dim1 lo
    g1[3] = (int)((BTILE >> 16) | ((unsigned)IDIM << 16));   // dim1 hi | tile_dim0=128
    g1[4] = (int)BTILE;                      // tile_dim1=32, tile_dim2=0
    g1[5] = (int)IDIM;                       // tensor_dim0_stride = 128 (lo32)
    g1[6] = 0;                               // stride hi | dim1_stride lo
    g1[7] = 0;
    i32x4 z4 = {0, 0, 0, 0};
#if defined(__clang_major__) && (__clang_major__ >= 23)
    i32x8 z8 = {0, 0, 0, 0, 0, 0, 0, 0};
    __builtin_amdgcn_tensor_load_to_lds(g0, g1, z4, z4, z8, 0);
#else
    __builtin_amdgcn_tensor_load_to_lds(g0, g1, z4, z4, 0);
#endif
}
#endif

extern "C" __global__ __launch_bounds__(256, 1)
void vecattgru_wmma(const float* __restrict__ xin,
                    const float* __restrict__ att,
                    const int*   __restrict__ slen,
                    const float* __restrict__ Wg,
                    const float* __restrict__ bg,
                    const float* __restrict__ Wc,
                    const float* __restrict__ bc,
                    float*       __restrict__ out)
{
    extern __shared__ char smem[];
    __bf16* sWg = (__bf16*)(smem + OFF_WG);   // 256 x WSTR
    __bf16* sWc = (__bf16*)(smem + OFF_WC);   // 128 x WSTR
    __bf16* sXH = (__bf16*)(smem + OFF_XH);   // BTILE x WSTR  [x | h] bf16
    __bf16* sRH = (__bf16*)(smem + OFF_RH);   // BTILE x RHSTR r*h bf16
    float*  sH  = (float*) (smem + OFF_H);    // BTILE x FSTR  h fp32
    float*  sU  = (float*) (smem + OFF_U);    // BTILE x FSTR  (1-a)*u fp32
    float*  sBG = (float*) (smem + OFF_BG);
    float*  sBC = (float*) (smem + OFF_BC);
    float*  sA  = (float*) (smem + OFF_A);
    int*    sSL = (int*)   (smem + OFF_SL);

    const int tid  = threadIdx.x;
    const int lane = tid & 31;
    const int wave = tid >> 5;
    const int g    = lane >> 4;     // half-wave group
    const int n16  = lane & 15;
    const int m    = wave >> 2;     // M-tile (0..1): batch rows m*16..m*16+15
    const int q    = wave & 3;      // N-split (0..3)
    const int wgBase = blockIdx.x * BTILE;

#ifdef HAVE_TDM
    // kick off DMA of x_0 immediately; it overlaps the weight staging below
    if (wave == 0)
        tdm_load_x(xin + (size_t)wgBase * IDIM, OFF_STAGE);
#endif

    // ---- one-time staging: weights fp32 -> bf16 LDS, biases, seq_lengths, h=0 ----
    for (int i4 = tid; i4 < (256*256)/4; i4 += 256) {
        int i = i4 * 4, row = i >> 8, col = i & 255;
        float4 f = ((const float4*)Wg)[i4];
        __bf16* d = sWg + row*WSTR + col;
        d[0]=(__bf16)f.x; d[1]=(__bf16)f.y; d[2]=(__bf16)f.z; d[3]=(__bf16)f.w;
    }
    for (int i4 = tid; i4 < (128*256)/4; i4 += 256) {
        int i = i4 * 4, row = i >> 8, col = i & 255;
        float4 f = ((const float4*)Wc)[i4];
        __bf16* d = sWc + row*WSTR + col;
        d[0]=(__bf16)f.x; d[1]=(__bf16)f.y; d[2]=(__bf16)f.z; d[3]=(__bf16)f.w;
    }
    if (tid < 256)  sBG[tid] = bg[tid];
    if (tid < 128)  sBC[tid] = bc[tid];
    if (tid < BTILE) sSL[tid] = slen[wgBase + tid];
    for (int i = tid; i < BTILE*HDIM; i += 256) {
        int row = i >> 7, col = i & 127;
        sH[row*FSTR + col]        = 0.0f;
        sXH[row*WSTR + 128 + col] = (__bf16)0.0f;
    }
    __syncthreads();

    const int xrow = tid >> 3;         // 0..31
    const int xcol = (tid & 7) * 16;   // 16 floats per thread

    for (int t = 0; t < T_STEPS; ++t) {
        // -------- phase A: make x_t available as bf16 in sXH, stage a_t --------
#ifdef HAVE_TDM
        if (wave == 0)
            __builtin_amdgcn_s_wait_tensorcnt((short)0);   // x_t DMA complete
        __syncthreads();
        if (wave == 0 && t + 1 < T_STEPS)                  // prefetch x_{t+1} via TDM
            tdm_load_x(xin + ((size_t)(t+1)*BATCH + wgBase)*IDIM,
                       OFF_STAGE + (unsigned)((t+1) & 1) * STAGE_BYTES);
        {
            const float* st = (const float*)(smem + OFF_STAGE + (unsigned)(t & 1)*STAGE_BYTES);
            const float* src = st + xrow*IDIM + xcol;
            __bf16* dst = sXH + xrow*WSTR + xcol;
            #pragma unroll
            for (int j = 0; j < 4; ++j) {
                float4 f = ((const float4*)src)[j];
                dst[4*j+0]=(__bf16)f.x; dst[4*j+1]=(__bf16)f.y;
                dst[4*j+2]=(__bf16)f.z; dst[4*j+3]=(__bf16)f.w;
            }
            if (tid < BTILE) sA[tid] = att[(size_t)t*BATCH + wgBase + tid];
        }
#else
        {
            const float* src = xin + ((size_t)t*BATCH + wgBase + xrow)*IDIM + xcol;
            __bf16* dst = sXH + xrow*WSTR + xcol;
            #pragma unroll
            for (int j = 0; j < 4; ++j) {
                float4 f = ((const float4*)src)[j];
                dst[4*j+0]=(__bf16)f.x; dst[4*j+1]=(__bf16)f.y;
                dst[4*j+2]=(__bf16)f.z; dst[4*j+3]=(__bf16)f.w;
            }
            if (t + 1 < T_STEPS)
                __builtin_prefetch(src + (size_t)BATCH*IDIM, 0, 0);
            if (tid < BTILE) sA[tid] = att[(size_t)t*BATCH + wgBase + tid];
        }
#endif
        __syncthreads();

        // -------- phase B: gates  ru = sigmoid(xh @ Wg^T + bg) --------
        {
            v16bf aF[8];
            const __bf16* aRow = sXH + (m*16 + n16)*WSTR;
            #pragma unroll
            for (int k = 0; k < 8; ++k) aF[k] = load_frag_a(aRow + k*32, g);

            #pragma unroll
            for (int i = 0; i < 4; ++i) {
                const int ntile = q*4 + i;          // 0..15 (r: 0..7, u: 8..15)
                v8f acc = {};
                const __bf16* bBase = sWg + (ntile*16 + n16)*WSTR + 16*g;
                #pragma unroll
                for (int k = 0; k < 8; ++k) {
                    v16bf bF = load_frag_b(bBase + k*32);
                    acc = __builtin_amdgcn_wmma_f32_16x16x32_bf16(
                        false, aF[k], false, bF, (short)0, acc, false, false);
                }
                const int n = ntile*16 + n16;
                const float bgn = sBG[n];
                #pragma unroll
                for (int r = 0; r < 8; ++r) {
                    const int row = m*16 + r + 8*g;
                    float s = sigmoidf_fast(acc[r] + bgn);
                    if (ntile < 8) {                     // reset gate -> stage r*h (bf16)
                        float h = sH[row*FSTR + n];
                        sRH[row*RHSTR + n] = (__bf16)(s * h);
                    } else {                             // update gate -> (1-a)*u (fp32)
                        const int uc = n - 128;
                        sU[row*FSTR + uc] = (1.0f - sA[row]) * s;
                    }
                }
            }
        }
        __syncthreads();

        // -------- phase C: candidate c = tanh([x, r*h] @ Wc^T + bc), state update ----
        {
            v16bf aF[8];
            const __bf16* aRowX = sXH + (m*16 + n16)*WSTR;
            const __bf16* aRowR = sRH + (m*16 + n16)*RHSTR;
            #pragma unroll
            for (int k = 0; k < 4; ++k) aF[k]     = load_frag_a(aRowX + k*32, g);
            #pragma unroll
            for (int k = 0; k < 4; ++k) aF[4 + k] = load_frag_a(aRowR + k*32, g);

            #pragma unroll
            for (int i = 0; i < 2; ++i) {
                const int ntc = q*2 + i;               // 0..7
                v8f acc = {};
                const __bf16* bBase = sWc + (ntc*16 + n16)*WSTR + 16*g;
                #pragma unroll
                for (int k = 0; k < 8; ++k) {
                    v16bf bF = load_frag_b(bBase + k*32);
                    acc = __builtin_amdgcn_wmma_f32_16x16x32_bf16(
                        false, aF[k], false, bF, (short)0, acc, false, false);
                }
                const int n = ntc*16 + n16;
                const float bcn = sBC[n];
                #pragma unroll
                for (int r = 0; r < 8; ++r) {
                    const int row = m*16 + r + 8*g;
                    float c  = tanhf_fast(acc[r] + bcn);
                    float u  = sU[row*FSTR + n];
                    float h  = sH[row*FSTR + n];
                    float nh = u*h + (1.0f - u)*c;
                    bool  msk = sSL[row] > t;
                    __builtin_nontemporal_store(msk ? nh : 0.0f,
                        &out[((size_t)t*BATCH + wgBase + row)*HDIM + n]);
                    float hn = msk ? nh : h;
                    sH[row*FSTR + n]        = hn;
                    sXH[row*WSTR + 128 + n] = (__bf16)hn;
                }
            }
        }
        __syncthreads();
    }

    // -------- final hidden state hx --------
    const size_t TBH = (size_t)T_STEPS * BATCH * HDIM;
    for (int i = tid; i < BTILE*HDIM; i += 256) {
        int row = i >> 7, col = i & 127;
        out[TBH + (size_t)(wgBase + row)*HDIM + col] = sH[row*FSTR + col];
    }
}

extern "C" void kernel_launch(void* const* d_in, const int* in_sizes, int n_in,
                              void* d_out, int out_size, void* d_ws, size_t ws_size,
                              hipStream_t stream) {
    (void)in_sizes; (void)n_in; (void)out_size; (void)d_ws; (void)ws_size;
    const float* xin = (const float*)d_in[0];   // inputs      (T, B, I) fp32
    const float* att = (const float*)d_in[1];   // att_scores  (T, B, 1) fp32
    const int*   sl  = (const int*)  d_in[2];   // seq_lengths (B,)      int32
    const float* Wg  = (const float*)d_in[3];   // (2H, I+H)
    const float* bg  = (const float*)d_in[4];   // (2H,)
    const float* Wc  = (const float*)d_in[5];   // (H, I+H)
    const float* bc  = (const float*)d_in[6];   // (H,)
    float* out = (float*)d_out;                 // outputs (T,B,H) ++ hx (B,H)

    vecattgru_wmma<<<dim3(NWG), dim3(256), SMEM_BYTES, stream>>>(
        xin, att, sl, Wg, bg, Wc, bc, out);
}
